// FNO4d_56745107915073
// MI455X (gfx1250) — matrix operand order repaired and checked by reading, not compiled, not run on hardware
//
#include <hip/hip_runtime.h>
#include <hip/hip_bf16.h>
#include <math.h>

typedef __attribute__((ext_vector_type(16))) __bf16 v16bf;
typedef __attribute__((ext_vector_type(8)))  __bf16 v8bf;
typedef __attribute__((ext_vector_type(8)))  float  v8f;

#define XD 64
#define YD 64
#define ZD 32
#define NP (XD*YD*ZD)      // 131072 spatial points
#define CC 96
#define TWO_PI 6.283185307179586f

// ---------------------------------------------------------------------------
// Twiddle tables (device-computed):
//   Tz  [8][32]  : e^{-2*pi*i * k*z/32},  k = 0..7
//   Txy [16][64] : e^{-2*pi*i * k*y/64},  k in {0..7, 56..63}  (kept modes)
// ---------------------------------------------------------------------------
__global__ void k_init_twiddles(float2* __restrict__ Tz, float2* __restrict__ Txy) {
  int t = threadIdx.x;
  if (t < 16 * 64) {
    int ki = t >> 6, y = t & 63;
    int k = (ki < 8) ? ki : (48 + ki);   // 8..15 -> 56..63
    float ang = -TWO_PI * (float)(k * y) / 64.0f;
    float s, c; __sincosf(ang, &s, &c);
    Txy[t] = make_float2(c, s);
  }
  if (t < 8 * 32) {
    int k = t >> 5, z = t & 31;
    float ang = -TWO_PI * (float)(k * z) / 32.0f;
    float s, c; __sincosf(ang, &s, &c);
    Tz[t] = make_float2(c, s);
  }
}

// ---------------------------------------------------------------------------
// Pack a weight matrix into WMMA A-fragment order (bf16), zero-padded.
// Fragment map (A 16x32 bf16, wave32): frag = (mt*ksteps + ks)*32 + lane,
//   m = mt*16 + (lane&15); elem j -> k = ks*32 + 16*(lane>>4) + j
// ---------------------------------------------------------------------------
__global__ void k_pack_A(const float* __restrict__ A, int sa_m, int sa_k,
                         int Mtot, int Kvalid, int ksteps,
                         __bf16* __restrict__ P, int frags) {
  int t = blockIdx.x * blockDim.x + threadIdx.x;
  if (t >= frags) return;
  int lane = t & 31;
  int ks = (t >> 5) % ksteps;
  int mt = (t >> 5) / ksteps;
  int g = lane >> 4;
  int m = mt * 16 + (lane & 15);
  __bf16* out = P + (size_t)t * 16;
#pragma unroll
  for (int j = 0; j < 16; ++j) {
    int k = ks * 32 + 16 * g + j;
    float v = (m < Mtot && k < Kvalid) ? A[(size_t)m * sa_m + (size_t)k * sa_k] : 0.0f;
    out[j] = (__bf16)v;
  }
}

// ---------------------------------------------------------------------------
// Build v (bf16, point-major [NP][32]): 25 transposed vars | gx gy gz | 0 pad
// x_in: [1,64,64,32,5(t),5(var)];  feature f = var*5 + t
// ---------------------------------------------------------------------------
__global__ void k_pack_v(const float* __restrict__ xin, __bf16* __restrict__ vb) {
  int p = blockIdx.x * blockDim.x + threadIdx.x;
  if (p >= NP) return;
  int z = p & 31, y = (p >> 5) & 63, x = p >> 11;
  const float* xp = xin + (size_t)p * 25;
  __bf16* vp = vb + (size_t)p * 32;
#pragma unroll
  for (int var = 0; var < 5; ++var)
#pragma unroll
    for (int t5 = 0; t5 < 5; ++t5)
      vp[var * 5 + t5] = (__bf16)xp[t5 * 5 + var];
  vp[25] = (__bf16)((float)x * (1.0f / 63.0f));
  vp[26] = (__bf16)((float)y * (1.0f / 63.0f));
  vp[27] = (__bf16)((float)z * (1.0f / 31.0f));
#pragma unroll
  for (int f = 28; f < 32; ++f) vp[f] = (__bf16)0.0f;
}

// ---------------------------------------------------------------------------
// Pointwise GEMM, bf16 WMMA, f32 accumulate. One wave = 1 m-tile x 4 p-tiles.
//   Apack : fragment-packed weights (k_pack_A)
//   Bb    : bf16 activations, point-major [NP][KP]; lane's 16 B elems are a
//           contiguous 32B chunk (B layout: n=lane&15, K=j+16*(lane>>4))
// Epilogue: +bias, +residual S (f32 [M][NP]), exact GELU, then any of:
//   OutF f32 [M][NP] channel-major,  OutB bf16 [NP][KPo] point-major (one
//   16B v8bf store per tile per lane), OutP f32 point-major [NP][Mtot].
// ---------------------------------------------------------------------------
template<int KSTEPS>
__global__ __launch_bounds__(256)
void k_gemm(const __bf16* __restrict__ Apack,
            const __bf16* __restrict__ Bb, int KP,
            const float* __restrict__ bias,
            const float* __restrict__ S,
            float* __restrict__ OutF,
            __bf16* __restrict__ OutB, int KPo,
            float* __restrict__ OutP,
            int Mtot, int mtiles, int apply_gelu) {
  const int lane = threadIdx.x & 31;
  const int wave = blockIdx.x * 8 + (threadIdx.x >> 5);
  const int pgroups = NP / 64;                 // 4 p-tiles per wave
  if (wave >= pgroups * mtiles) return;        // wave-uniform
  const int tm = wave / pgroups;
  const int p0 = (wave % pgroups) * 64;
  const int g = lane >> 4, nn = lane & 15;
  const int m0 = tm * 16;

  v8f acc[4] = {{}, {}, {}, {}};
  const __bf16* ap = Apack + ((size_t)tm * KSTEPS * 32 + lane) * 16;
  const __bf16* bp0 = Bb + (size_t)(p0 + nn) * KP + 16 * g;
#pragma unroll
  for (int ks = 0; ks < KSTEPS; ++ks) {
    v16bf a = *(const v16bf*)(ap + (size_t)ks * 512);
    const __bf16* bp = bp0 + ks * 32;
    v16bf b0 = *(const v16bf*)(bp);
    v16bf b1 = *(const v16bf*)(bp + (size_t)16 * KP);
    v16bf b2 = *(const v16bf*)(bp + (size_t)32 * KP);
    v16bf b3 = *(const v16bf*)(bp + (size_t)48 * KP);
    acc[0] = __builtin_amdgcn_wmma_f32_16x16x32_bf16(false, a, false, b0, (short)0, acc[0], false, false);
    acc[1] = __builtin_amdgcn_wmma_f32_16x16x32_bf16(false, a, false, b1, (short)0, acc[1], false, false);
    acc[2] = __builtin_amdgcn_wmma_f32_16x16x32_bf16(false, a, false, b2, (short)0, acc[2], false, false);
    acc[3] = __builtin_amdgcn_wmma_f32_16x16x32_bf16(false, a, false, b3, (short)0, acc[3], false, false);
  }

  float bv[8];
#pragma unroll
  for (int r = 0; r < 8; ++r) bv[r] = bias ? bias[m0 + 8 * g + r] : 0.0f;

#pragma unroll
  for (int t = 0; t < 4; ++t) {
    const int p = p0 + t * 16 + nn;
    v8bf packed;
#pragma unroll
    for (int r = 0; r < 8; ++r) {
      const int m = m0 + 8 * g + r;
      float vv = acc[t][r] + bv[r];
      if (S) vv += S[(size_t)m * NP + p];
      if (apply_gelu) vv = 0.5f * vv * (1.0f + erff(vv * 0.70710678118f));
      if (OutF) OutF[(size_t)m * NP + p] = vv;
      if (OutP && m < Mtot) OutP[(size_t)p * Mtot + m] = vv;
      packed[r] = (__bf16)vv;
    }
    if (OutB) *(v8bf*)(OutB + (size_t)p * KPo + m0 + 8 * g) = packed;
  }
}

// ---------------------------------------------------------------------------
// Truncated separable forward DFT: Z (real->8 modes), Y (64->16), X (64->16)
// ---------------------------------------------------------------------------
__global__ void k_dft_z(const float* __restrict__ h, float2* __restrict__ Fz,
                        const float2* __restrict__ Tz) {
  size_t t = (size_t)blockIdx.x * blockDim.x + threadIdx.x;
  if (t >= (size_t)CC * XD * YD * 8) return;
  int kz = t & 7; size_t row = t >> 3;
  const float* hp = h + row * ZD;
  float re = 0.f, im = 0.f;
#pragma unroll 4
  for (int z = 0; z < ZD; ++z) {
    float v = hp[z]; float2 w = Tz[kz * 32 + z];
    re += v * w.x; im += v * w.y;
  }
  Fz[row * 8 + kz] = make_float2(re, im);
}

__global__ void k_dft_y(const float2* __restrict__ Fz, float2* __restrict__ Fyz,
                        const float2* __restrict__ Txy) {
  size_t t = (size_t)blockIdx.x * blockDim.x + threadIdx.x;
  if (t >= (size_t)CC * XD * 16 * 8) return;
  int kz = t & 7, ky = (t >> 3) & 15; size_t cx = t >> 7;
  const float2* fp = Fz + cx * (size_t)YD * 8 + kz;
  float re = 0.f, im = 0.f;
#pragma unroll 4
  for (int y = 0; y < YD; ++y) {
    float2 a = fp[(size_t)y * 8]; float2 w = Txy[ky * 64 + y];
    re += a.x * w.x - a.y * w.y;
    im += a.x * w.y + a.y * w.x;
  }
  Fyz[(cx * 16 + ky) * 8 + kz] = make_float2(re, im);
}

__global__ void k_dft_x(const float2* __restrict__ Fyz, float2* __restrict__ F,
                        const float2* __restrict__ Txy) {
  size_t t = (size_t)blockIdx.x * blockDim.x + threadIdx.x;
  if (t >= (size_t)CC * 16 * 16 * 8) return;
  int kz = t & 7, ky = (t >> 3) & 15, kx = (t >> 7) & 15; int c = t >> 11;
  float re = 0.f, im = 0.f;
#pragma unroll 4
  for (int x = 0; x < XD; ++x) {
    float2 a = Fyz[(((size_t)c * XD + x) * 16 + ky) * 8 + kz];
    float2 w = Txy[kx * 64 + x];
    re += a.x * w.x - a.y * w.y;
    im += a.x * w.y + a.y * w.x;
  }
  F[(((size_t)c * 16 + kx) * 16 + ky) * 8 + kz] = make_float2(re, im);
}

// ---------------------------------------------------------------------------
// Per-mode complex channel mix:  G[o,mode] = sum_i F[i,mode] * Wc[corner,i,o,mode]
// w_spec: [4,4,96,96,8,8,8,2].  Block = one mode (2048), thread = o (96).
// ---------------------------------------------------------------------------
__global__ __launch_bounds__(96)
void k_mode_mix(const float2* __restrict__ F, const float* __restrict__ wspec,
                float2* __restrict__ G, int layer) {
  int mm = blockIdx.x;                 // (kx*16 + ky)*8 + kz
  int o  = threadIdx.x;
  int kz = mm & 7, ky = (mm >> 3) & 15, kx = mm >> 7;
  int corner = (kx >= 8 ? 1 : 0) + (ky >= 8 ? 2 : 0);
  int moff = ((kx & 7) * 8 + (ky & 7)) * 8 + kz;

  __shared__ float2 Fs[CC];
  Fs[o] = F[(size_t)o * 2048 + mm];
  __syncthreads();

  size_t base = (((size_t)(layer * 4 + corner) * 9216 + o) * 512 + moff) * 2;
  const size_t istride = (size_t)96 * 512 * 2;
  float re = 0.f, im = 0.f;
  for (int i = 0; i < CC; ++i) {
    const float* wp = wspec + base + (size_t)i * istride;
    float wr = wp[0], wi = wp[1];
    float2 a = Fs[i];
    re += a.x * wr - a.y * wi;
    im += a.x * wi + a.y * wr;
  }
  G[(size_t)o * 2048 + mm] = make_float2(re, im);
}

// ---------------------------------------------------------------------------
// Inverse stages (conjugate twiddles); final Z stage real w/ Hermitian fold,
// carries the full 1/(X*Y*Z) normalization.
// ---------------------------------------------------------------------------
__global__ void k_inv_x(const float2* __restrict__ G, float2* __restrict__ Gx,
                        const float2* __restrict__ Txy) {
  size_t t = (size_t)blockIdx.x * blockDim.x + threadIdx.x;
  if (t >= (size_t)CC * XD * 16 * 8) return;
  int kz = t & 7, ky = (t >> 3) & 15, x = (t >> 7) & 63; int c = t >> 13;
  float re = 0.f, im = 0.f;
#pragma unroll
  for (int kx = 0; kx < 16; ++kx) {
    float2 a = G[(((size_t)c * 16 + kx) * 16 + ky) * 8 + kz];
    float2 w = Txy[kx * 64 + x];           // conj
    re += a.x * w.x + a.y * w.y;
    im += a.y * w.x - a.x * w.y;
  }
  Gx[(((size_t)c * XD + x) * 16 + ky) * 8 + kz] = make_float2(re, im);
}

__global__ void k_inv_y(const float2* __restrict__ Gx, float2* __restrict__ Gxy,
                        const float2* __restrict__ Txy) {
  size_t t = (size_t)blockIdx.x * blockDim.x + threadIdx.x;
  if (t >= (size_t)CC * XD * YD * 8) return;
  int kz = t & 7, y = (t >> 3) & 63, x = (t >> 9) & 63; int c = t >> 15;
  float re = 0.f, im = 0.f;
#pragma unroll
  for (int ky = 0; ky < 16; ++ky) {
    float2 a = Gx[(((size_t)c * XD + x) * 16 + ky) * 8 + kz];
    float2 w = Txy[ky * 64 + y];           // conj
    re += a.x * w.x + a.y * w.y;
    im += a.y * w.x - a.x * w.y;
  }
  Gxy[(((size_t)c * XD + x) * YD + y) * 8 + kz] = make_float2(re, im);
}

__global__ void k_inv_z(const float2* __restrict__ Gxy, float* __restrict__ s1,
                        const float2* __restrict__ Tz) {
  size_t t = (size_t)blockIdx.x * blockDim.x + threadIdx.x;
  if (t >= (size_t)CC * NP) return;
  int z = t & 31; size_t row = t >> 5;
  const float2* gp = Gxy + row * 8;
  float acc = gp[0].x;
#pragma unroll
  for (int k = 1; k < 8; ++k) {
    float2 a = gp[k];
    float2 w = Tz[k * 32 + z];             // (cos,-sin): Re(a*e^{+i th}) = a.x*w.x + a.y*w.y
    acc += 2.0f * (a.x * w.x + a.y * w.y);
  }
  s1[t] = acc * (1.0f / (float)NP);
}

// ---------------------------------------------------------------------------
// Host-side orchestration
// ---------------------------------------------------------------------------
template<int KSTEPS>
static void launch_gemm(const __bf16* Apack, const __bf16* Bb, int KP,
                        const float* bias, const float* S,
                        float* OutF, __bf16* OutB, int KPo, float* OutP,
                        int Mtot, int mtiles, int gelu, hipStream_t stream) {
  int waves = (NP / 64) * mtiles;
  k_gemm<KSTEPS><<<waves / 8, 256, 0, stream>>>(Apack, Bb, KP, bias, S, OutF,
                                                OutB, KPo, OutP, Mtot, mtiles, gelu);
}

static void launch_pack_A(const float* A, int sa_m, int sa_k, int Mtot, int Kvalid,
                          int mtiles, int ksteps, __bf16* P, hipStream_t stream) {
  int frags = mtiles * ksteps * 32;
  k_pack_A<<<(frags + 255) / 256, 256, 0, stream>>>(A, sa_m, sa_k, Mtot, Kvalid,
                                                    ksteps, P, frags);
}

extern "C" void kernel_launch(void* const* d_in, const int* in_sizes, int n_in,
                              void* d_out, int out_size, void* d_ws, size_t ws_size,
                              hipStream_t stream) {
  const float* x_in  = (const float*)d_in[0];
  const float* wspec = (const float*)d_in[1];
  const float* wskip = (const float*)d_in[2];
  const float* bskip = (const float*)d_in[3];
  const float* fc0w  = (const float*)d_in[4];
  const float* fc0b  = (const float*)d_in[5];
  const float* fc1w  = (const float*)d_in[6];
  const float* fc1b  = (const float*)d_in[7];
  const float* fc2w  = (const float*)d_in[8];
  const float* fc2b  = (const float*)d_in[9];
  float* out = (float*)d_out;
  (void)in_sizes; (void)n_in; (void)out_size; (void)ws_size;

  char* ws = (char*)d_ws;
  size_t off = 0;
  auto alloc = [&](size_t bytes) -> void* {
    void* p = ws + off; off += (bytes + 255) & ~(size_t)255; return p;
  };
  float*   h   = (float*)alloc(sizeof(float) * CC * NP);            // 50.3 MB  f32 ch-major
  __bf16*  hb0 = (__bf16*)alloc(sizeof(__bf16) * (size_t)NP * CC);  // 25.2 MB
  __bf16*  hb1 = (__bf16*)alloc(sizeof(__bf16) * (size_t)NP * CC);  // 25.2 MB
  float*   s1  = (float*)alloc(sizeof(float) * CC * NP);            // 50.3 MB
  float2*  Fz  = (float2*)alloc(sizeof(float2) * CC * XD * YD * 8); // 25.2 MB
  float2*  Fyz = (float2*)alloc(sizeof(float2) * CC * XD * 16 * 8); // 6.3 MB
  float2*  Fk  = (float2*)alloc(sizeof(float2) * CC * 2048);        // 1.6 MB
  float2*  Gk  = (float2*)alloc(sizeof(float2) * CC * 2048);        // 1.6 MB
  float2*  Tz  = (float2*)alloc(sizeof(float2) * 8 * 32);
  float2*  Txy = (float2*)alloc(sizeof(float2) * 16 * 64);
  __bf16*  Afc0 = (__bf16*)alloc(sizeof(__bf16) * 6 * 1 * 512);
  __bf16*  Askp[4];
  for (int l = 0; l < 4; ++l) Askp[l] = (__bf16*)alloc(sizeof(__bf16) * 6 * 3 * 512);
  __bf16*  Afc1 = (__bf16*)alloc(sizeof(__bf16) * 8 * 3 * 512);
  __bf16*  Afc2 = (__bf16*)alloc(sizeof(__bf16) * 1 * 4 * 512);
  // Aliases over dead regions (s1 first written at layer0 inv_z; dead after layer3):
  __bf16* vb    = (__bf16*)s1;   // [NP][32] bf16 (8.4 MB), used only before layer 0
  __bf16* t128b = (__bf16*)s1;   // [NP][128] bf16 (33.6 MB), used only after layers

  k_init_twiddles<<<1, 1024, 0, stream>>>(Tz, Txy);
  k_pack_v<<<NP / 256, 256, 0, stream>>>(x_in, vb);

  // Weight packs (A(m,k) indexing): fc0_w[k*96+m], w_skip[l][m*96+k],
  // fc1_w[k*128+m], fc2_w[k*5+m]
  launch_pack_A(fc0w, 1, CC, CC, 28, 6, 1, Afc0, stream);
  for (int l = 0; l < 4; ++l)
    launch_pack_A(wskip + (size_t)l * CC * CC, CC, 1, CC, CC, 6, 3, Askp[l], stream);
  launch_pack_A(fc1w, 1, 128, 128, CC, 8, 3, Afc1, stream);
  launch_pack_A(fc2w, 1, 5, 5, 128, 1, 4, Afc2, stream);

  // fc0: h (f32) + hb0 (bf16) = v @ fc0_w + fc0_b
  launch_gemm<1>(Afc0, vb, 32, fc0b, nullptr, h, hb0, CC, nullptr, CC, 6, 0, stream);

  for (int l = 0; l < 4; ++l) {
    __bf16* hbin  = (l & 1) ? hb1 : hb0;
    __bf16* hbout = (l & 1) ? hb0 : hb1;
    k_dft_z<<<(CC * XD * YD * 8) / 256, 256, 0, stream>>>(h, Fz, Tz);
    k_dft_y<<<(CC * XD * 16 * 8) / 256, 256, 0, stream>>>(Fz, Fyz, Txy);
    k_dft_x<<<(CC * 16 * 16 * 8) / 256, 256, 0, stream>>>(Fyz, Fk, Txy);
    k_mode_mix<<<2048, 96, 0, stream>>>(Fk, wspec, Gk, l);
    k_inv_x<<<(CC * XD * 16 * 8) / 256, 256, 0, stream>>>(Gk, Fyz, Txy);
    k_inv_y<<<(CC * XD * YD * 8) / 256, 256, 0, stream>>>(Fyz, Fz, Txy);
    k_inv_z<<<(CC * NP) / 256, 256, 0, stream>>>(Fz, s1, Tz);
    // h/hbout = gelu?( s1 + hin @ w_skip[l]^T + b_skip[l] )
    launch_gemm<3>(Askp[l], hbin, CC, bskip + (size_t)l * CC, s1,
                   h, hbout, CC, nullptr, CC, 6, (l < 3) ? 1 : 0, stream);
  }

  // fc1: t128b = gelu(h @ fc1_w + fc1_b)   (activations end in hb0)
  launch_gemm<3>(Afc1, hb0, CC, fc1b, nullptr, nullptr, t128b, 128, nullptr,
                 128, 8, 1, stream);
  // fc2: out[p][5] = t128b @ fc2_w + fc2_b
  launch_gemm<4>(Afc2, t128b, 128, fc2b, nullptr, nullptr, nullptr, 0, out,
                 5, 1, 0, stream);
}
